// HEAT_21955872817582
// MI455X (gfx1250) — compile-verified, matrix operand order checked by reading, not run on hardware
//
#include <hip/hip_runtime.h>

#define NEG_SLOPE 0.2f

typedef __attribute__((ext_vector_type(16))) __bf16 v16bf;
typedef __attribute__((ext_vector_type(8)))  __bf16 v8bf;
typedef __attribute__((ext_vector_type(8)))  float  v8f;

__device__ __forceinline__ float leaky(float v) { return v > 0.f ? v : NEG_SLOPE * v; }
__device__ __forceinline__ __bf16 f2bf(float f) { return (__bf16)f; }
// K-offset pattern for 16-bit A/B fragments of v_wmma_*_16x16x32 (CDNA5 ISA 7.12.2):
// lane<16 covers K {0..7,16..23}, lane>=16 covers K {8..15,24..31}; VGPR v holds a bf16 pair.
__device__ __forceinline__ int kpat(int v, int hi) { return (v < 4 ? 2 * v : 16 + 2 * (v - 4)) + hi * 8; }

// A fragment from a row-major bf16 row: two contiguous 16B runs (k ks..ks+7, ks+16..ks+23).
// p must already include row base + hi*8 + chunk offset.
__device__ __forceinline__ v16bf load_a16(const __bf16* p) {
  v8bf lo = *(const v8bf*)p;
  v8bf hh = *(const v8bf*)(p + 16);
  v16bf a;
#pragma unroll
  for (int i = 0; i < 8; ++i) { a[i] = lo[i]; a[8 + i] = hh[i]; }
  return a;
}

__device__ __forceinline__ v8f wmma_bf16(v16bf a, v16bf b, v8f c) {
  return __builtin_amdgcn_wmma_f32_16x16x32_bf16(false, a, false, b, (short)0, c, false, false);
}

// monotonic float <-> ordered-uint mapping so atomicMax(u32) implements float max
__device__ __forceinline__ unsigned ford(float f) {
  unsigned u = __float_as_uint(f);
  return (u & 0x80000000u) ? ~u : (u | 0x80000000u);
}
__device__ __forceinline__ float funord(unsigned u) {
  return __uint_as_float((u & 0x80000000u) ? (u & 0x7fffffffu) : ~u);
}

// ------------- weight pre-swizzle: fp32 [K][N] (or transposed [N][K]) -> bf16 WMMA B fragments ---
// frag layout: [kchunk][ntile][lane 0..31][16 bf16]; one thread emits one lane's 16 elements.
__global__ void k_prep_frag(const float* __restrict__ W, __bf16* __restrict__ frag,
                            int kchunks, int ntiles, int ldw, int kvalid, int trans) {
  int idx = blockIdx.x * blockDim.x + threadIdx.x;
  int total = kchunks * ntiles * 32;
  if (idx >= total) return;
  int l = idx & 31;
  int t = (idx >> 5) % ntiles;
  int c = (idx >> 5) / ntiles;
  int m = l & 15, hi = l >> 4;
  int n = t * 16 + m;
  __bf16* o = frag + (size_t)idx * 16;
#pragma unroll
  for (int v = 0; v < 8; ++v) {
#pragma unroll
    for (int odd = 0; odd < 2; ++odd) {
      int k = c * 32 + kpat(v, hi) + odd;
      float val = 0.f;
      if (k < kvalid) val = trans ? W[(size_t)n * ldw + k] : W[(size_t)k * ldw + n];
      o[2 * v + odd] = f2bf(val);
    }
  }
}

// ------------- fp32 -> bf16 buffer mirror ----------------
__global__ void k_f2bf(const float* __restrict__ in, __bf16* __restrict__ out, size_t n) {
  size_t i = (size_t)blockIdx.x * blockDim.x + threadIdx.x;
  size_t st = (size_t)gridDim.x * blockDim.x;
  for (; i < n; i += st) out[i] = f2bf(in[i]);
}

// ------------- node-wise dense linear: out = act(xb[n,64] @ W + b), W pre-swizzled bf16 ---------
// grid: ceil(n/16); blockDim = (NCOLS/16)*32
template <int NCOLS, bool RELU>
__global__ __launch_bounds__(128) void k_node_linear(
    const __bf16* __restrict__ xb, const __bf16* __restrict__ wfrag,
    const float* __restrict__ bias, float* __restrict__ out,
    __bf16* __restrict__ outb, int n) {
  constexpr int NT = NCOLS / 16;
  int base = blockIdx.x * 16;
  int wt = threadIdx.x >> 5;
  int l = threadIdx.x & 31, m = l & 15, hi = l >> 4;
  int row = base + m; if (row >= n) row = n - 1;
  const __bf16* ar = xb + (size_t)row * 64 + hi * 8;
  v8f acc = {};
#pragma unroll
  for (int c = 0; c < 2; ++c) {
    v16bf a = load_a16(ar + c * 32);
    v16bf b = *(const v16bf*)(wfrag + ((size_t)(c * NT + wt) * 32 + l) * 16);
    acc = wmma_bf16(a, b, acc);
  }
  float bv = bias[wt * 16 + m];
  if (base + 16 <= n) {                       // uniform fast path: whole tile in range
    float*  p  = out  + (size_t)(base + hi * 8) * NCOLS + wt * 16 + m;
    __bf16* pb = outb ? outb + (size_t)(base + hi * 8) * NCOLS + wt * 16 + m : (__bf16*)0;
#pragma unroll
    for (int r = 0; r < 8; ++r) {
      float vv = acc[r] + bv;
      if (RELU) vv = fmaxf(vv, 0.f);
      p[r * NCOLS] = vv;
      if (outb) pb[r * NCOLS] = f2bf(vv);
    }
  } else {
#pragma unroll
    for (int r = 0; r < 8; ++r) {
      int mr = base + r + hi * 8;
      if (mr < n) {
        float vv = acc[r] + bv;
        if (RELU) vv = fmaxf(vv, 0.f);
        out[(size_t)mr * NCOLS + wt * 16 + m] = vv;
        if (outb) outb[(size_t)mr * NCOLS + wt * 16 + m] = f2bf(vv);
      }
    }
  }
}

// ------------- HeteroLinear: 3 per-type 64x64 GEMMs, branch-free per-row type select ------------
__global__ __launch_bounds__(128) void k_hetero(
    const __bf16* __restrict__ xb, const __bf16* __restrict__ wfrag, // [3][2][4][32][16]
    const float* __restrict__ hb, const int* __restrict__ node_type,
    float* __restrict__ xh, __bf16* __restrict__ xhb, int n) {
  int base = blockIdx.x * 16;
  int wt = threadIdx.x >> 5;
  int l = threadIdx.x & 31, m = l & 15, hi = l >> 4;
  int row = base + m; if (row >= n) row = n - 1;
  const __bf16* ar = xb + (size_t)row * 64 + hi * 8;
  v16bf a0 = load_a16(ar);
  v16bf a1 = load_a16(ar + 32);
  int tr[8];
#pragma unroll
  for (int r = 0; r < 8; ++r) {
    int mr = base + r + hi * 8; if (mr >= n) mr = n - 1;
    tr[r] = node_type[mr];
  }
  float res[8] = {};
#pragma unroll
  for (int t = 0; t < 3; ++t) {
    const __bf16* wf = wfrag + (size_t)t * (2 * 4 * 32 * 16);
    v16bf b0 = *(const v16bf*)(wf + ((size_t)(0 * 4 + wt) * 32 + l) * 16);
    v16bf b1 = *(const v16bf*)(wf + ((size_t)(1 * 4 + wt) * 32 + l) * 16);
    v8f acc = {};
    acc = wmma_bf16(a0, b0, acc);
    acc = wmma_bf16(a1, b1, acc);
    float bv = hb[t * 64 + wt * 16 + m];
#pragma unroll
    for (int r = 0; r < 8; ++r) res[r] = (tr[r] == t) ? acc[r] + bv : res[r];
  }
  if (base + 16 <= n) {
    float*  p  = xh  + (size_t)(base + hi * 8) * 64 + wt * 16 + m;
    __bf16* pb = xhb + (size_t)(base + hi * 8) * 64 + wt * 16 + m;
#pragma unroll
    for (int r = 0; r < 8; ++r) { p[r * 64] = res[r]; pb[r * 64] = f2bf(res[r]); }
  } else {
#pragma unroll
    for (int r = 0; r < 8; ++r) {
      int mr = base + r + hi * 8;
      if (mr < n) {
        xh [(size_t)mr * 64 + wt * 16 + m] = res[r];
        xhb[(size_t)mr * 64 + wt * 16 + m] = f2bf(res[r]);
      }
    }
  }
}

// ------------- init softmax accumulators / zero ----------------
__global__ void k_init_softmax(unsigned* __restrict__ amax, float* __restrict__ den, int n) {
  int i = blockIdx.x * blockDim.x + threadIdx.x;
  if (i < n) { amax[i] = 0x007FFFFFu; /* ford(-inf) */ den[i] = 0.f; }
}

__global__ void k_zero(float* __restrict__ p, size_t n) {
  size_t i = (size_t)blockIdx.x * blockDim.x + threadIdx.x;
  size_t st = (size_t)gridDim.x * blockDim.x;
  for (; i < n; i += st) p[i] = 0.f;
}

// ------------- per-edge: edge-attr embedding, attention logits, running segment max -------------
__global__ void k_alpha(const float* __restrict__ xh, const float* __restrict__ edge_attr,
                        const float* __restrict__ ete_tbl, const float* __restrict__ eaW,
                        const float* __restrict__ attW, const int* __restrict__ ei,
                        const int* __restrict__ edge_type,
                        float* __restrict__ ea_out, float* __restrict__ alpha_out,
                        unsigned* __restrict__ amax, int E) {
  int e = blockIdx.x * blockDim.x + threadIdx.x;
  if (e >= E) return;
  int src = ei[e], dst = ei[E + e];
  float d0 = edge_attr[2 * (size_t)e], d1 = edge_attr[2 * (size_t)e + 1];
  float eav[6];
#pragma unroll
  for (int j = 0; j < 6; ++j) {
    eav[j] = leaky(eaW[2 * j] * d0 + eaW[2 * j + 1] * d1);
    ea_out[6 * (size_t)e + j] = eav[j];
  }
  float etev[5];
  int et = edge_type[e];
#pragma unroll
  for (int i = 0; i < 5; ++i) etev[i] = leaky(ete_tbl[5 * et + i]);
  const float* xi = xh + (size_t)dst * 64;
  const float* xj = xh + (size_t)src * 64;
#pragma unroll
  for (int h = 0; h < 2; ++h) {
    const float* w = attW + h * 139;
    float s = 0.f;
#pragma unroll 8
    for (int k = 0; k < 64; ++k) s += xi[k] * w[k];
#pragma unroll 8
    for (int k = 0; k < 64; ++k) s += xj[k] * w[64 + k];
#pragma unroll
    for (int i = 0; i < 5; ++i) s += etev[i] * w[128 + i];
#pragma unroll
    for (int j = 0; j < 6; ++j) s += eav[j] * w[133 + j];
    float al = leaky(s);
    alpha_out[2 * (size_t)e + h] = al;
    atomicMax(&amax[2 * (size_t)dst + h], ford(al));
  }
}

// ------------- exp(alpha - max) in place + segment denominator ----------------
__global__ void k_expsum(float* __restrict__ alf, const unsigned* __restrict__ amax,
                         const int* __restrict__ ei, float* __restrict__ den, int E) {
  int e = blockIdx.x * blockDim.x + threadIdx.x;
  if (e >= E) return;
  int dst = ei[E + e];
#pragma unroll
  for (int h = 0; h < 2; ++h) {
    float mx = funord(amax[2 * (size_t)dst + h]);
    float ex = __expf(alf[2 * (size_t)e + h] - mx);
    alf[2 * (size_t)e + h] = ex;
    atomicAdd(&den[2 * (size_t)dst + h], ex);
  }
}

// ------------- per-edge scalar = mean over heads of softmax weight ----------------
__global__ void k_scale(const float* __restrict__ ex, const float* __restrict__ den,
                        const int* __restrict__ ei, float* __restrict__ scl, int E) {
  int e = blockIdx.x * blockDim.x + threadIdx.x;
  if (e >= E) return;
  int dst = ei[E + e];
  float s0 = ex[2 * (size_t)e]     / (den[2 * (size_t)dst]     + 1e-16f);
  float s1 = ex[2 * (size_t)e + 1] / (den[2 * (size_t)dst + 1] + 1e-16f);
  scl[e] = 0.5f * (s0 + s1);
}

// ------------- message GEMM [16 edges,70pad96]@[96,64] via WMMA + scaled atomic scatter ---------
__global__ __launch_bounds__(128) void k_msg(
    const __bf16* __restrict__ xhb, const float* __restrict__ ea,
    const float* __restrict__ scl, const __bf16* __restrict__ wfrag, // [3][4][32][16], K padded
    const int* __restrict__ ei, float* __restrict__ out, int E) {
  int e0 = blockIdx.x * 16;
  int wt = threadIdx.x >> 5;
  int l = threadIdx.x & 31, m = l & 15, hi = l >> 4;
  int er = e0 + m; if (er >= E) er = E - 1;
  int src = ei[er];
  const __bf16* ar = xhb + (size_t)src * 64 + hi * 8;   // gathered bf16 row (L2-resident)
  const float* earow = ea + (size_t)er * 6;
  v8f acc = {};
#pragma unroll
  for (int c = 0; c < 2; ++c) {
    v16bf a = load_a16(ar + c * 32);
    v16bf b = *(const v16bf*)(wfrag + ((size_t)(c * 4 + wt) * 32 + l) * 16);
    acc = wmma_bf16(a, b, acc);
  }
  { // K chunk [64,96): 6 edge-attr-emb values then zero padding (B frag already zero-padded)
    v16bf a;
#pragma unroll
    for (int v = 0; v < 8; ++v) {
      int kb = kpat(v, hi);
      a[2 * v]     = (kb     < 6) ? f2bf(earow[kb])     : f2bf(0.f);
      a[2 * v + 1] = (kb + 1 < 6) ? f2bf(earow[kb + 1]) : f2bf(0.f);
    }
    v16bf b = *(const v16bf*)(wfrag + ((size_t)(2 * 4 + wt) * 32 + l) * 16);
    acc = wmma_bf16(a, b, acc);
  }
  if (e0 + 16 <= E) {
#pragma unroll
    for (int r = 0; r < 8; ++r) {
      int e = e0 + hi * 8 + r;
      int dst = ei[E + e];
      atomicAdd(&out[(size_t)dst * 64 + wt * 16 + m], acc[r] * scl[e]);
    }
  } else {
#pragma unroll
    for (int r = 0; r < 8; ++r) {
      int e = e0 + hi * 8 + r;
      if (e < E) {
        int dst = ei[E + e];
        atomicAdd(&out[(size_t)dst * 64 + wt * 16 + m], acc[r] * scl[e]);
      }
    }
  }
}

static inline size_t rnd8(size_t v) { return (v + 7) & ~(size_t)7; }

extern "C" void kernel_launch(void* const* d_in, const int* in_sizes, int n_in,
                              void* d_out, int out_size, void* d_ws, size_t ws_size,
                              hipStream_t stream) {
  const float* x         = (const float*)d_in[0];
  const float* edge_attr = (const float*)d_in[1];
  const float* W_in      = (const float*)d_in[2];
  const float* b_in      = (const float*)d_in[3];
  const float* hetero_W  = (const float*)d_in[4];
  const float* hetero_b  = (const float*)d_in[5];
  const float* ete_tbl   = (const float*)d_in[6];
  const float* eaW       = (const float*)d_in[7];
  const float* attW      = (const float*)d_in[8];
  const float* msgW      = (const float*)d_in[9];
  const float* W_out     = (const float*)d_in[10];
  const float* b_out     = (const float*)d_in[11];
  const int*   ei        = (const int*)d_in[12];
  const int*   node_type = (const int*)d_in[13];
  const int*   edge_type = (const int*)d_in[14];

  int N = in_sizes[0] / 64;
  int E = in_sizes[14];
  const int L = 2, T = 3;

  float* ws = (float*)d_ws;
  size_t o = 0;
  float* hA  = ws + o; o += rnd8((size_t)N * 64);
  float* hB  = ws + o; o += rnd8((size_t)N * 64);
  float* xh  = ws + o; o += rnd8((size_t)N * 64);
  float* ea  = ws + o; o += rnd8((size_t)E * 6);
  float* alf = ws + o; o += rnd8((size_t)E * 2);
  float* scl = ws + o; o += rnd8((size_t)E);
  unsigned* amax = (unsigned*)(ws + o); o += rnd8((size_t)N * 2);
  float* den = ws + o; o += rnd8((size_t)N * 2);
  // bf16 mirrors + weight fragments (32B-aligned: all offsets multiples of 8 floats)
  __bf16* xbin = (__bf16*)(ws + o); o += rnd8((size_t)N * 32);
  __bf16* hAb  = (__bf16*)(ws + o); o += rnd8((size_t)N * 32);
  __bf16* hBb  = (__bf16*)(ws + o); o += rnd8((size_t)N * 32);
  __bf16* xhb  = (__bf16*)(ws + o); o += rnd8((size_t)N * 32);
  const int FRAG = 2 * 4 * 32 * 16;          // 4096 bf16 per 64x64 weight
  __bf16* fr_in  = (__bf16*)(ws + o); o += rnd8(FRAG / 2);
  __bf16* fr_het = (__bf16*)(ws + o); o += rnd8((size_t)L * T * FRAG / 2);
  const int MFRAG = 3 * 4 * 32 * 16;         // 6144 bf16 per msgW (K=70 pad 96)
  __bf16* fr_msg = (__bf16*)(ws + o); o += rnd8((size_t)L * MFRAG / 2);
  const int OFRAG = 2 * 2 * 32 * 16;         // 2048 bf16 for W_out (64x32)
  __bf16* fr_out = (__bf16*)(ws + o); o += rnd8(OFRAG / 2);

  int NT = (N + 15) / 16;
  int ET = (E + 15) / 16;
  int EB = (E + 255) / 256;

  // ---- weight pre-swizzle (tiny) ----
  k_prep_frag<<<1, 256, 0, stream>>>(W_in, fr_in, 2, 4, 64, 64, 0);
  for (int l = 0; l < L; ++l)
    for (int t = 0; t < T; ++t)
      k_prep_frag<<<1, 256, 0, stream>>>(hetero_W + (size_t)(l * T + t) * 64 * 64,
                                         fr_het + (size_t)(l * T + t) * FRAG, 2, 4, 64, 64, 0);
  for (int l = 0; l < L; ++l)
    k_prep_frag<<<2, 256, 0, stream>>>(msgW + (size_t)l * 64 * 70,
                                       fr_msg + (size_t)l * MFRAG, 3, 4, 70, 70, 1);
  k_prep_frag<<<1, 128, 0, stream>>>(W_out, fr_out, 2, 2, 32, 64, 0);

  // ---- input linear + relu (writes fp32 + bf16 mirror) ----
  k_f2bf<<<512, 256, 0, stream>>>(x, xbin, (size_t)N * 64);
  k_node_linear<64, true><<<NT, 128, 0, stream>>>(xbin, fr_in, b_in, hA, hAb, N);

  float*  hin  = hA;  float*  hout  = hB;
  __bf16* hinb = hAb; __bf16* houtb = hBb;
  for (int l = 0; l < L; ++l) {
    k_hetero<<<NT, 128, 0, stream>>>(hinb, fr_het + (size_t)l * T * FRAG,
                                     hetero_b + (size_t)l * T * 64, node_type, xh, xhb, N);
    k_init_softmax<<<(N * 2 + 255) / 256, 256, 0, stream>>>(amax, den, N * 2);
    k_alpha<<<EB, 256, 0, stream>>>(xh, edge_attr, ete_tbl + (size_t)l * 4 * 5,
                                    eaW + (size_t)l * 12, attW + (size_t)l * 2 * 139,
                                    ei, edge_type, ea, alf, amax, E);
    k_expsum<<<EB, 256, 0, stream>>>(alf, amax, ei, den, E);
    k_scale<<<EB, 256, 0, stream>>>(alf, den, ei, scl, E);
    k_zero<<<1024, 256, 0, stream>>>(hout, (size_t)N * 64);
    k_msg<<<ET, 128, 0, stream>>>(xhb, ea, scl, fr_msg + (size_t)l * MFRAG, ei, hout, E);
    k_f2bf<<<512, 256, 0, stream>>>(hout, houtb, (size_t)N * 64);
    float* tf = hin; hin = hout; hout = tf;
    __bf16* tb = hinb; hinb = houtb; houtb = tb;
  }

  // ---- output linear (64 -> 32), 2 col-tiles -> 2 waves ----
  k_node_linear<32, false><<<NT, 64, 0, stream>>>(hinb, fr_out, b_out, (float*)d_out,
                                                  (__bf16*)0, N);
}
